// SpatialAttention_64106681860662
// MI455X (gfx1250) — matrix operand order repaired and checked by reading, not compiled, not run on hardware
//
#include <hip/hip_runtime.h>

// MI455X / gfx1250 flash-attention for SpatialAttention reference.
// Phase 1: pack_qkv  — fuse per-channel scale+bias, head-major relayout, f32->bf16.
//                      Q folded with d^-0.5 * log2(e); V packed TRANSPOSED [bh][d][hw].
// Phase 2: flash_attn — wave32 WMMA flash attention computed TRANSPOSED:
//                      S^T = K·Q^T and O^T = V^T·P^T (each lane owns ONE query:
//                      scalar m/l state, one cross-half permlane exchange),
//                      K/V tiles double-buffered via TDM tensor_load_to_lds
//                      (hardware-padded LDS rows) or async global->LDS fallback.
//
// Workspace: 3 * 4*4*4096*64 * 2 = 24 MiB.

typedef __attribute__((ext_vector_type(16))) __bf16       v16bf;
typedef __attribute__((ext_vector_type(8)))  float        v8f;
typedef __attribute__((ext_vector_type(4)))  int          v4i;
typedef __attribute__((ext_vector_type(8)))  int          v8i;
typedef __attribute__((ext_vector_type(4)))  unsigned int v4u;

#define BATCH   4
#define HEADS   4
#define SEQ     4096
#define HDIM    64
#define QTILE   128      // q rows per block (8 waves x 16)
#define KTILE   64       // keys per iteration
#define NITER   (SEQ / KTILE)
#define NWAVES  8
#define LPAD    72       // padded LDS row stride in ushorts (144B: conflict-free, 16B aligned)

#if defined(__has_builtin)
#  if __has_builtin(__builtin_amdgcn_global_load_async_to_lds_b128)
#    define HAVE_ASYNC 1
#  endif
#  if __has_builtin(__builtin_amdgcn_permlanex16)
#    define HAVE_PERMX 1
#  endif
#  if __has_builtin(__builtin_amdgcn_tensor_load_to_lds)
#    define HAVE_TDM 1
#  endif
#endif
#ifndef HAVE_ASYNC
#  define HAVE_ASYNC 0
#endif
#ifndef HAVE_PERMX
#  define HAVE_PERMX 0
#endif
#ifndef HAVE_TDM
#  define HAVE_TDM 0
#endif

union Frag {
    unsigned int u32[8];
    v16bf        bf;
};

__device__ __forceinline__ unsigned short f32_to_bf16(float f) {
    unsigned int u = __float_as_uint(f);
    u += 0x7FFFu + ((u >> 16) & 1u);        // round-to-nearest-even
    return (unsigned short)(u >> 16);
}

// exchange value with the other 16-lane half (lane ^ 16), VALU permlane
__device__ __forceinline__ float xhalf(float x) {
#if HAVE_PERMX
    const int xi = __float_as_int(x);
    return __int_as_float(__builtin_amdgcn_permlanex16(
        xi, xi, 0x76543210, 0xfedcba98, false, false));
#else
    return __shfl_xor(x, 16, 32);
#endif
}

// ---------------------------------------------------------------------------
// Phase 1: elementwise prologue + relayout. 1,048,576 threads, 4 elems each.
// ---------------------------------------------------------------------------
__global__ __launch_bounds__(256) void pack_qkv(
    const float* __restrict__ q_in, const float* __restrict__ k_in,
    const float* __restrict__ v_in,
    const float* __restrict__ wq, const float* __restrict__ bq,
    const float* __restrict__ wk, const float* __restrict__ bk,
    const float* __restrict__ wv, const float* __restrict__ bv,
    unsigned short* __restrict__ qp, unsigned short* __restrict__ kp,
    unsigned short* __restrict__ vpT)
{
    const float QSCALE = 0.125f * 1.44269504088896340f;   // d^-0.5 * log2(e)
    const int t   = blockIdx.x * 256 + threadIdx.x;       // 0 .. 1048575
    const int e   = t * 4;                                // flat (b,hw,c) element idx
    const int c   = e & 255;
    const int bhw = e >> 8;                               // b*4096 + hw
    const int hd  = c >> 6, dc = c & 63;
    const int b   = bhw >> 12, hw = bhw & 4095;
    const int bh  = b * HEADS + hd;
    const size_t dst = ((size_t)(bh * SEQ + hw)) * HDIM + dc;

    union Pack4 { unsigned short s[4]; unsigned long long u; };

    {   // Q (head-major row layout)
        const float4 x = *(const float4*)(q_in + e);
        const float4 w = *(const float4*)(wq + c);
        const float4 bi = *(const float4*)(bq + c);
        Pack4 o;
        o.s[0] = f32_to_bf16((x.x * w.x + bi.x) * QSCALE);
        o.s[1] = f32_to_bf16((x.y * w.y + bi.y) * QSCALE);
        o.s[2] = f32_to_bf16((x.z * w.z + bi.z) * QSCALE);
        o.s[3] = f32_to_bf16((x.w * w.w + bi.w) * QSCALE);
        *(unsigned long long*)(qp + dst) = o.u;
    }
    {   // K (head-major row layout)
        const float4 x = *(const float4*)(k_in + e);
        const float4 w = *(const float4*)(wk + c);
        const float4 bi = *(const float4*)(bk + c);
        Pack4 o;
        o.s[0] = f32_to_bf16(x.x * w.x + bi.x);
        o.s[1] = f32_to_bf16(x.y * w.y + bi.y);
        o.s[2] = f32_to_bf16(x.z * w.z + bi.z);
        o.s[3] = f32_to_bf16(x.w * w.w + bi.w);
        *(unsigned long long*)(kp + dst) = o.u;
    }
    {   // V, stored transposed: vpT[bh][d][hw]
        const float4 x = *(const float4*)(v_in + e);
        const float4 w = *(const float4*)(wv + c);
        const float4 bi = *(const float4*)(bv + c);
        const size_t vbase = ((size_t)(bh * HDIM + dc)) * SEQ + hw;
        vpT[vbase + 0 * SEQ] = f32_to_bf16(x.x * w.x + bi.x);
        vpT[vbase + 1 * SEQ] = f32_to_bf16(x.y * w.y + bi.y);
        vpT[vbase + 2 * SEQ] = f32_to_bf16(x.z * w.z + bi.z);
        vpT[vbase + 3 * SEQ] = f32_to_bf16(x.w * w.w + bi.w);
    }
}

// ---------------------------------------------------------------------------
// TDM: one descriptor fetches a 64x64 bf16 tile into LDS with hardware row
// padding (pad 4 DWORDs after every 32 DWORDs -> 144-byte LDS rows == LPAD).
// ---------------------------------------------------------------------------
#if HAVE_TDM
__device__ __forceinline__ void tdm_load_tile(
    const unsigned short* gaddr, unsigned lds_addr,
    unsigned tensor_dim0, unsigned tensor_dim1, unsigned stride0)
{
    const unsigned long long ga = (unsigned long long)gaddr;
    v4u g0;
    g0[0] = 1u;                                          // count=1, user descriptor
    g0[1] = lds_addr;                                    // LDS byte address
    g0[2] = (unsigned)(ga & 0xFFFFFFFFu);                // global addr [31:0]
    g0[3] = (unsigned)((ga >> 32) & 0x01FFFFFFu)         // global addr [56:32]
          | (2u << 30);                                  // type = 2 (image/tensor)
    v8i g1;
    g1[0] = (int)((1u << 16)                             // data_size = 2 bytes
                | (1u << 20)                             // pad_enable
                | (4u << 22)                             // pad_interval: 32 DWORDs (128B)
                | (3u << 25));                           // pad_amount:   4 DWORDs (16B)
    g1[1] = (int)(tensor_dim0 << 16);                    // tensor_dim0 [15:0] @ bits 63:48
    g1[2] = (int)((tensor_dim0 >> 16) | ((tensor_dim1 & 0xFFFFu) << 16));
    g1[3] = (int)((tensor_dim1 >> 16) | (64u << 16));    // tile_dim0 = 64
    g1[4] = 64;                                          // tile_dim1 = 64, tile_dim2 = 0
    g1[5] = (int)stride0;                                // tensor_dim0_stride [31:0]
    g1[6] = 0;
    g1[7] = 0;
    const v4i z4 = {0, 0, 0, 0};                         // groups 2/3: unused (2-D tile)
    const v8i z8 = {0, 0, 0, 0, 0, 0, 0, 0};             // trailing group: unused (2-D)
    __builtin_amdgcn_tensor_load_to_lds(g0, g1, z4, z4, z8, 0);
}
#endif

// ---------------------------------------------------------------------------
// Fallback tile loader: per-thread async (or sync) b128 copies.
// ---------------------------------------------------------------------------
__device__ __forceinline__ void load_tiles(
    const unsigned short* __restrict__ kbp,   // K base   [hw][d]  for this bh
    const unsigned short* __restrict__ vbpT,  // V^T base [d][hw]  for this bh
    int k0, int tid,
    unsigned short (*__restrict__ Kt)[LPAD],
    unsigned short (*__restrict__ Vt)[LPAD])
{
    #pragma unroll
    for (int i = 0; i < 2; ++i) {
        const int idx  = tid + i * 256;               // 0..511
        const int row  = idx >> 3;                    // tile row (key for K, d for V)
        const int quad = idx & 7;                     // 8-ushort (16B) chunk
        const unsigned short* gk = kbp  + (size_t)(k0 + row) * HDIM + quad * 8;
        const unsigned short* gv = vbpT + (size_t)row * SEQ + k0 + quad * 8;
        unsigned short* lk = &Kt[row][quad * 8];
        unsigned short* lv = &Vt[row][quad * 8];
#if HAVE_ASYNC
        typedef __attribute__((address_space(1))) v4i* as1v4;   // global int4*
        typedef __attribute__((address_space(3))) v4i* as3v4;   // LDS int4*
        __builtin_amdgcn_global_load_async_to_lds_b128(
            (as1v4)(unsigned long long)gk,
            (as3v4)(unsigned)(unsigned long long)lk, 0, 0);
        __builtin_amdgcn_global_load_async_to_lds_b128(
            (as1v4)(unsigned long long)gv,
            (as3v4)(unsigned)(unsigned long long)lv, 0, 0);
#else
        *(uint4*)lk = *(const uint4*)gk;
        *(uint4*)lv = *(const uint4*)gv;
#endif
    }
}

// ---------------------------------------------------------------------------
// Phase 2: flash attention (transposed). 512 blocks x 256 threads (8 wave32).
// ---------------------------------------------------------------------------
__global__ __launch_bounds__(256) void flash_attn(
    const unsigned short* __restrict__ qp, const unsigned short* __restrict__ kp,
    const unsigned short* __restrict__ vpT,
    const float* __restrict__ wp, const float* __restrict__ bp,
    float* __restrict__ out)
{
    __shared__ unsigned short Kt[2][KTILE][LPAD];         // K tile [key][d], double buffered
    __shared__ unsigned short Vt[2][HDIM ][LPAD];         // V^T tile [d][key], double buffered
    __shared__ unsigned short Pq[NWAVES][16][LPAD];       // per-wave P staging, P[q][key]

    const int tid  = threadIdx.x;
    const int wave = tid >> 5;
    const int lane = tid & 31;
    const int hf   = lane >> 4;                           // half-wave select
    const int ln   = lane & 15;

    const int bh    = blockIdx.x >> 5;                    // 0..15  (b*4 + h)
    const int qt    = blockIdx.x & 31;
    const int qbase = qt * QTILE + wave * 16;
    const int head  = bh & 3;
    const int batch = bh >> 2;

    // --- Q strip (16 q x 64 d) as B-fragments (Q^T), 2 chunks of K=32 ---
    // B layout: lane (hf,ln): column n = q = ln; elements k = d = 32*kc + 16*hf + 2v,2v+1
    Frag bq[2];
    {
        const unsigned short* qrow = qp + ((size_t)(bh * SEQ + qbase + ln)) * HDIM;
        #pragma unroll
        for (int kc = 0; kc < 2; ++kc)
            #pragma unroll
            for (int v = 0; v < 8; ++v)
                bq[kc].u32[v] = *(const unsigned int*)(qrow + kc * 32 + 16 * hf + 2 * v);
    }

    const v8f vzero = {0.f,0.f,0.f,0.f,0.f,0.f,0.f,0.f};
    v8f   oacc[4];                                        // O^T: d-tile nt, lane: d=nt*16+r+8hf, q=ln
    float mrow = -3.0e38f, lrow = 0.f;                    // scalar per lane (one query each)
    #pragma unroll
    for (int nt = 0; nt < 4; ++nt) oacc[nt] = vzero;

    const unsigned short* kbp  = kp  + (size_t)bh * SEQ * HDIM;
    const unsigned short* vbpT = vpT + (size_t)bh * HDIM * SEQ;

    // --- prologue: prefetch tile 0 ---
#if HAVE_TDM
    if (wave == 0) {
        tdm_load_tile(kbp,  (unsigned)(unsigned long long)&Kt[0][0][0], HDIM, KTILE, HDIM);
        tdm_load_tile(vbpT, (unsigned)(unsigned long long)&Vt[0][0][0], SEQ,  HDIM,  SEQ);
        __builtin_amdgcn_s_wait_tensorcnt(0);
    }
#else
    load_tiles(kbp, vbpT, 0, tid, Kt[0], Vt[0]);
#  if HAVE_ASYNC
    asm volatile("s_wait_asynccnt 0" ::: "memory");
#  endif
#endif
    __syncthreads();

    for (int it = 0; it < NITER; ++it) {
        const int cur = it & 1;

        // prefetch next tile into the other buffer (overlaps with compute)
#if HAVE_TDM
        if (wave == 0 && it + 1 < NITER) {
            const int k1 = (it + 1) * KTILE;
            tdm_load_tile(kbp + (size_t)k1 * HDIM,
                          (unsigned)(unsigned long long)&Kt[cur ^ 1][0][0],
                          HDIM, KTILE, HDIM);
            tdm_load_tile(vbpT + k1,
                          (unsigned)(unsigned long long)&Vt[cur ^ 1][0][0],
                          SEQ, HDIM, SEQ);
        }
#else
        if (it + 1 < NITER)
            load_tiles(kbp, vbpT, (it + 1) * KTILE, tid, Kt[cur ^ 1], Vt[cur ^ 1]);
#endif

        // --- S^T = K * Q^T : 4 key-tiles x 16 q per wave ---
        // A layout: lane (hf,ln): row m = key = t*16+ln; k = d = 32*kc + kb
        v8f s[4];
        #pragma unroll
        for (int t = 0; t < 4; ++t) s[t] = vzero;
        #pragma unroll
        for (int kc = 0; kc < 2; ++kc) {
            #pragma unroll
            for (int t = 0; t < 4; ++t) {
                Frag ak;
                #pragma unroll
                for (int v = 0; v < 8; ++v) {
                    const int kb = 16 * hf + 2 * v;
                    ak.u32[v] = *(const unsigned int*)&Kt[cur][t * 16 + ln][kc * 32 + kb];
                }
                s[t] = __builtin_amdgcn_wmma_f32_16x16x32_bf16(
                           false, ak.bf, false, bq[kc].bf, (short)0, s[t], false, false);
            }
        }
        // lane (hf,ln) now holds scores for q=ln, keys { t*16 + 8*hf + r }.

        // --- online softmax (base-2; per-lane scalar state, one cross-half hop) ---
        float mx = -3.0e38f;
        #pragma unroll
        for (int t = 0; t < 4; ++t)
            #pragma unroll
            for (int r = 0; r < 8; ++r) mx = fmaxf(mx, s[t][r]);
        mx = fmaxf(mx, xhalf(mx));
        const float mnew  = fmaxf(mrow, mx);
        const float alpha = __builtin_amdgcn_exp2f(mrow - mnew);
        lrow *= alpha;
        #pragma unroll
        for (int nt = 0; nt < 4; ++nt)
            #pragma unroll
            for (int r = 0; r < 8; ++r) oacc[nt][r] *= alpha;

        float psum = 0.f;
        #pragma unroll
        for (int t = 0; t < 4; ++t) {
            union { __bf16 h[8]; uint4 q; } pk;
            #pragma unroll
            for (int r = 0; r < 8; ++r) {
                const float p = __builtin_amdgcn_exp2f(s[t][r] - mnew);
                psum += p;
                pk.h[r] = (__bf16)p;
            }
            // 8 consecutive keys of row q=ln -> one b128 store
            *(uint4*)&Pq[wave][ln][t * 16 + 8 * hf] = pk.q;
        }
        lrow += psum + xhalf(psum);
        mrow  = mnew;

        // --- O^T += V^T * P^T  (A = V^T tiles, B = P^T shared across d-tiles) ---
        #pragma unroll
        for (int kc = 0; kc < 2; ++kc) {
            Frag bpf;      // B: lane (hf,ln): col n=q=ln, k = key = 32*kc + 16*hf + 2v
            #pragma unroll
            for (int v = 0; v < 8; ++v)
                bpf.u32[v] = *(const unsigned int*)&Pq[wave][ln][kc * 32 + 16 * hf + 2 * v];
            #pragma unroll
            for (int nt = 0; nt < 4; ++nt) {
                Frag av;   // A: lane (hf,ln): row m = d = nt*16+ln; k = key = 32*kc + kb
                #pragma unroll
                for (int v = 0; v < 8; ++v) {
                    const int kb = 16 * hf + 2 * v;
                    av.u32[v] = *(const unsigned int*)&Vt[cur][nt * 16 + ln][kc * 32 + kb];
                }
                oacc[nt] = __builtin_amdgcn_wmma_f32_16x16x32_bf16(
                               false, av.bf, false, bpf.bf, (short)0, oacc[nt], false, false);
            }
        }

        // publish the prefetched buffer
#if HAVE_TDM
        if (wave == 0) __builtin_amdgcn_s_wait_tensorcnt(0);
#else
#  if HAVE_ASYNC
        asm volatile("s_wait_asynccnt 0" ::: "memory");
#  endif
#endif
        __syncthreads();
    }

    // --- epilogue: normalize, apply wp/bp, vectorized f32 stores ---
    {
        const float inv = 1.0f / lrow;
        const int q = qbase + ln;                         // each lane owns one query row
        float* orow = out + (size_t)(batch * SEQ + q) * 256 + head * HDIM;
        const float* wrow = wp + head * HDIM;
        const float* brow = bp + head * HDIM;
        #pragma unroll
        for (int nt = 0; nt < 4; ++nt) {
            const int c0 = nt * 16 + 8 * hf;              // 8 consecutive channels
            const float4 w0 = *(const float4*)(wrow + c0);
            const float4 w1 = *(const float4*)(wrow + c0 + 4);
            const float4 b0 = *(const float4*)(brow + c0);
            const float4 b1 = *(const float4*)(brow + c0 + 4);
            float4 o0, o1;
            o0.x = oacc[nt][0] * inv * w0.x + b0.x;
            o0.y = oacc[nt][1] * inv * w0.y + b0.y;
            o0.z = oacc[nt][2] * inv * w0.z + b0.z;
            o0.w = oacc[nt][3] * inv * w0.w + b0.w;
            o1.x = oacc[nt][4] * inv * w1.x + b1.x;
            o1.y = oacc[nt][5] * inv * w1.y + b1.y;
            o1.z = oacc[nt][6] * inv * w1.z + b1.z;
            o1.w = oacc[nt][7] * inv * w1.w + b1.w;
            *(float4*)(orow + c0)     = o0;
            *(float4*)(orow + c0 + 4) = o1;
        }
    }
}

// ---------------------------------------------------------------------------
extern "C" void kernel_launch(void* const* d_in, const int* in_sizes, int n_in,
                              void* d_out, int out_size, void* d_ws, size_t ws_size,
                              hipStream_t stream) {
    const float* q_in = (const float*)d_in[0];
    const float* k_in = (const float*)d_in[1];
    const float* v_in = (const float*)d_in[2];
    const float* wq   = (const float*)d_in[3];
    const float* bq   = (const float*)d_in[4];
    const float* wk   = (const float*)d_in[5];
    const float* bk   = (const float*)d_in[6];
    const float* wv   = (const float*)d_in[7];
    const float* bv   = (const float*)d_in[8];
    const float* wp   = (const float*)d_in[9];
    const float* bp   = (const float*)d_in[10];

    const size_t NPACK = (size_t)BATCH * HEADS * SEQ * HDIM;   // 4,194,304 elems
    unsigned short* qp  = (unsigned short*)d_ws;
    unsigned short* kp  = qp + NPACK;
    unsigned short* vpT = kp + NPACK;

    pack_qkv<<<4096, 256, 0, stream>>>(q_in, k_in, v_in, wq, bq, wk, bk, wv, bv,
                                       qp, kp, vpT);
    flash_attn<<<BATCH * HEADS * (SEQ / QTILE), 256, 0, stream>>>(
        qp, kp, vpT, wp, bp, (float*)d_out);
}
